// OrientedAnchoredRSConv_9663676416048
// MI455X (gfx1250) — matrix operand order repaired and checked by reading, not compiled, not run on hardware
//
#include <hip/hip_runtime.h>

// ---------------------------------------------------------------------------
// OrientedAnchoredRSConv fused implementation for MI455X (gfx1250, wave32).
//   B=16, N_IN=4096, N_OUT=1024, K=32, F=4, C_IN=128, C_OUT=256, C4=32
// Roofline: ~6.5 GFLOP total; h_in (33.5MB) is L2-resident (192MB L2) so the
// per-edge channel gather is L2-bandwidth bound; HBM traffic ~60MB -> ~3us at
// 23.3 TB/s. All channel-mixing matmuls use v_wmma_f32_16x16x32_f16 (f16
// inputs, f32 accumulation). Fragments are built with wide ds_load_b128 from
// zero-padded LDS tiles (no lane-divergent predication).
// ---------------------------------------------------------------------------

#define B_      16
#define N_IN    4096
#define N_OUT   1024
#define K_NN    32
#define C_IN    128
#define C_OUT   256
#define C4_     32
#define EPSF    1e-5f

typedef __attribute__((ext_vector_type(16))) _Float16 v16h;
typedef __attribute__((ext_vector_type(8)))  _Float16 v8h;
typedef __attribute__((ext_vector_type(4)))  _Float16 v4h;
typedef __attribute__((ext_vector_type(8)))  float    v8f;

// ---- WMMA fragment loaders (ISA 7.12.2, 16-bit A 16x32 / B 32x16) ----------
// A (16x32): lanes 0-15 hold K 0..15, lanes 16-31 hold K 16..31 (matches b_frag)
// -> per lane: one contiguous v16h at half offset 16*(lane/16).
__device__ __forceinline__ v16h a_frag(const _Float16* rowp, int lane) {
    const int ab = (lane & 16) ? 16 : 0;
    return *(const v16h*)(rowp + ab);
}
// B (32x16) stored as [n][k] in LDS: lane = column n (mod 16); lanes 0-15 read
// K 0..15, lanes 16-31 read K 16..31 -> one contiguous v16h per lane.
__device__ __forceinline__ v16h b_frag(const _Float16* colp, int lane) {
    const int kb = (lane & 16) ? 16 : 0;
    return *(const v16h*)(colp + kb);
}

// ===========================================================================
// Kernel A: brute-force KNN, per-thread sorted top-32 list in LDS.
// ===========================================================================
#define KNN_T    128
#define KNN_TILE 512

__global__ void __launch_bounds__(KNN_T)
knn_kernel(const float* __restrict__ p_in, const float* __restrict__ p_out,
           int* __restrict__ idx_out) {
    __shared__ float sd[K_NN * KNN_T];
    __shared__ int   si[K_NN * KNN_T];
    __shared__ float sp[KNN_TILE * 3];

    const int tid = threadIdx.x;
    const int q   = blockIdx.x * KNN_T + tid;        // global query
    const int b   = (blockIdx.x * KNN_T) / N_OUT;    // uniform within block

    const float qx = p_out[q * 3 + 0];
    const float qy = p_out[q * 3 + 1];
    const float qz = p_out[q * 3 + 2];

#pragma unroll
    for (int j = 0; j < K_NN; ++j) { sd[j * KNN_T + tid] = 3.0e38f; si[j * KNN_T + tid] = 0; }
    float worst = 3.0e38f;

    for (int t0 = 0; t0 < N_IN; t0 += KNN_TILE) {
        __syncthreads();
        for (int i = tid; i < KNN_TILE * 3; i += KNN_T)
            sp[i] = p_in[b * N_IN * 3 + t0 * 3 + i];
        __syncthreads();
#pragma unroll 4
        for (int i = 0; i < KNN_TILE; ++i) {
            float dx = qx - sp[i * 3 + 0];
            float dy = qy - sp[i * 3 + 1];
            float dz = qz - sp[i * 3 + 2];
            float d  = dx * dx + dy * dy + dz * dz;
            if (d < worst) {
                int j = K_NN - 1;
                while (j > 0 && sd[(j - 1) * KNN_T + tid] > d) {
                    sd[j * KNN_T + tid] = sd[(j - 1) * KNN_T + tid];
                    si[j * KNN_T + tid] = si[(j - 1) * KNN_T + tid];
                    --j;
                }
                sd[j * KNN_T + tid] = d;
                si[j * KNN_T + tid] = t0 + i;
                worst = sd[(K_NN - 1) * KNN_T + tid];
            }
        }
    }
    __syncthreads();
#pragma unroll
    for (int j = 0; j < K_NN; ++j)
        idx_out[(size_t)q * K_NN + j] = si[j * KNN_T + tid];
}

// ===========================================================================
// Kernel B: per-edge geometry + MLP1 + MLP2 + weighted message + max over K.
// One workgroup (256 thr = 8 waves) per 8 output points (256 edges).
// ===========================================================================
#define PT   8              // points per tile
#define NE   (PT * K_NN)    // 256 edges per tile

__global__ void __launch_bounds__(256)
edge_kernel(const float* __restrict__ p_in,  const float* __restrict__ p_out,
            const float* __restrict__ R_out, const float* __restrict__ h_in,
            const float* __restrict__ w1,    const float* __restrict__ b1,
            const float* __restrict__ a_g,   const float* __restrict__ a_b,
            const float* __restrict__ a_m,   const float* __restrict__ a_v,
            const float* __restrict__ w2,    const float* __restrict__ b2,
            const float* __restrict__ bg,    const float* __restrict__ bb,
            const float* __restrict__ bm,    const float* __restrict__ bv,
            const int*   __restrict__ knn_idx,
            _Float16*    __restrict__ h_ws) {
    __shared__ int      s_idx[NE];
    __shared__ float    s_pout[PT * 3];
    __shared__ float    s_R[PT * 12 * 3];
    __shared__ __align__(64) _Float16 s_w1t[C4_ * 32];    // [n][k], k>=16 zero
    __shared__ __align__(64) _Float16 s_w2t[C_IN * 32];   // [c][k]
    __shared__ __align__(64) _Float16 s_wfeat[NE * 32];   // A of MLP1 (k>=16 zero)
    __shared__ __align__(64) _Float16 s_x[NE * 32];       // A of MLP2 (K=32 exact)
    __shared__ int      s_hi[PT * C_IN];                  // max-reduce (float bits)
    __shared__ float    s_b1[C4_], s_saA[C4_], s_shA[C4_];
    __shared__ float    s_b2[C_IN], s_scB[C_IN], s_shB[C_IN];

    const int tid  = threadIdx.x;
    const int lane = tid & 31;
    const int wave = tid >> 5;
    const int b    = blockIdx.x >> 7;        // 128 tiles per batch
    const int base = (blockIdx.x & 127) * PT;

    // ---------------- stage tile-constant data ----------------
    s_idx[tid] = knn_idx[((size_t)b * N_OUT + base) * K_NN + tid];
    if (tid < PT * 3)  s_pout[tid] = p_out[((size_t)b * N_OUT + base) * 3 + tid];
    for (int i = tid; i < PT * 36; i += 256)
        s_R[i] = R_out[((size_t)b * N_OUT + base) * 36 + i];
    for (int i = tid; i < C4_ * 16; i += 256) {      // w1 (C4,16) -> [n][k], pad
        int n = i >> 4, k = i & 15;
        s_w1t[n * 32 + k]      = (_Float16)w1[i];
        s_w1t[n * 32 + 16 + k] = (_Float16)0.f;
    }
    {   // w2 (C_IN,C4) row-major == [c][k]: straight f32->f16, float4 at a time
        const float4* w2v = (const float4*)w2;
        for (int i = tid; i < (C_IN * C4_) / 4; i += 256) {
            float4 w = w2v[i];
            v4h h = {(_Float16)w.x, (_Float16)w.y, (_Float16)w.z, (_Float16)w.w};
            *(v4h*)(s_w2t + i * 4) = h;
        }
    }
    if (tid < C4_) {
        float sc = a_g[tid] * rsqrtf(a_v[tid] + EPSF);
        s_saA[tid] = sc;
        s_shA[tid] = a_b[tid] - a_m[tid] * sc;
        s_b1[tid]  = b1[tid];
    }
    if (tid < C_IN) {
        float sc = bg[tid] * rsqrtf(bv[tid] + EPSF);
        s_scB[tid] = sc;
        s_shB[tid] = bb[tid] - bm[tid] * sc;
        s_b2[tid]  = b2[tid];
    }
    for (int i = tid; i < PT * C_IN; i += 256) s_hi[i] = 0;
    __syncthreads();

    // ---------------- geometry: one edge per thread ----------------
    {
        const int e  = tid;
        const int pl = e >> 5;
        const int nb = s_idx[e];
        // warm L2/L0 for the MLP2-phase channel gather
        __builtin_prefetch(h_in + ((size_t)b * N_IN + nb) * C_IN, 0, 1);
        const float px = p_in[(b * N_IN + nb) * 3 + 0];
        const float py = p_in[(b * N_IN + nb) * 3 + 1];
        const float pz = p_in[(b * N_IN + nb) * 3 + 2];
        const float rx = px - s_pout[pl * 3 + 0];
        const float ry = py - s_pout[pl * 3 + 1];
        const float rz = pz - s_pout[pl * 3 + 2];
        _Float16* row = s_wfeat + e * 32;
        float pij[12];
#pragma unroll
        for (int m = 0; m < 12; ++m) {
            pij[m] = s_R[pl * 36 + m * 3 + 0] * rx
                   + s_R[pl * 36 + m * 3 + 1] * ry
                   + s_R[pl * 36 + m * 3 + 2] * rz;
            row[m] = (_Float16)pij[m];
        }
#pragma unroll
        for (int f = 0; f < 4; ++f) {
            float d = sqrtf(pij[3 * f] * pij[3 * f] + pij[3 * f + 1] * pij[3 * f + 1]
                            + pij[3 * f + 2] * pij[3 * f + 2]);
            row[12 + f] = (_Float16)d;
        }
        v8h z = {};
        *(v8h*)(row + 16) = z;     // zero-pad K 16..31
        *(v8h*)(row + 24) = z;
    }
    __syncthreads();

    // ---------------- MLP1: (256x16 pad->32) @ (16x32) via WMMA ----------------
    for (int t = wave * 4; t < wave * 4 + 4; ++t) {
        const int mt = t >> 1, nt = t & 1;
        const int m  = mt * 16 + (lane & 15);
        const int j  = nt * 16 + (lane & 15);
        v16h a  = a_frag(s_wfeat + m * 32, lane);
        v16h bf = b_frag(s_w1t + j * 32, lane);
        v8f acc = {};
        acc = __builtin_amdgcn_wmma_f32_16x16x32_f16(false, a, false, bf,
                                                     (short)0, acc, false, false);
        const float sc = s_saA[j], sh = s_shA[j], bi = s_b1[j];
#pragma unroll
        for (int r = 0; r < 8; ++r) {
            int e = mt * 16 + r + ((lane >> 4) << 3);
            float v = fmaxf((acc[r] + bi) * sc + sh, 0.f);
            s_x[e * 32 + j] = (_Float16)v;
        }
    }
    __syncthreads();

    // ------- MLP2 (256x32)@(32x128) + h_j gather + BN/ReLU + max over K -------
    for (int t = wave; t < 128; t += 8) {
        const int mt = t >> 3, nt = t & 7;
        const int m  = mt * 16 + (lane & 15);
        const int c  = nt * 16 + (lane & 15);
        v16h a  = a_frag(s_x + m * 32, lane);
        v16h bf = b_frag(s_w2t + c * 32, lane);
        v8f acc = {};
        acc = __builtin_amdgcn_wmma_f32_16x16x32_f16(false, a, false, bf,
                                                     (short)0, acc, false, false);
        const float sc = s_scB[c], sh = s_shB[c], bi = s_b2[c];
#pragma unroll
        for (int r = 0; r < 8; ++r) {
            int e  = mt * 16 + r + ((lane >> 4) << 3);
            int pl = e >> 5;
            float hj = h_in[((size_t)b * N_IN + s_idx[e]) * C_IN + c]; // L2-resident
            float v  = fmaxf((acc[r] + bi) * hj * sc + sh, 0.f);
            atomicMax(&s_hi[pl * C_IN + c], __float_as_int(v));
        }
    }
    __syncthreads();

    // ---------------- write h (f16, packed v4h) to workspace ----------------
    {
        _Float16* dst = h_ws + ((size_t)b * N_OUT + base) * C_IN;
        const int i0 = tid * 4;   // PT*C_IN == 256*4
        v4h h = {(_Float16)__int_as_float(s_hi[i0 + 0]),
                 (_Float16)__int_as_float(s_hi[i0 + 1]),
                 (_Float16)__int_as_float(s_hi[i0 + 2]),
                 (_Float16)__int_as_float(s_hi[i0 + 3])};
        *(v4h*)(dst + i0) = h;
    }
}

// ===========================================================================
// Kernel C: out = relu(bn(h @ w3^T + b3)) ; 16384x256x128 WMMA GEMM.
// Block tile 128M x 64N, K loop 4 x 32.
// ===========================================================================
__global__ void __launch_bounds__(256)
out_gemm_kernel(const _Float16* __restrict__ h_ws, const float* __restrict__ w3,
                const float* __restrict__ b3,
                const float* __restrict__ cg, const float* __restrict__ cb,
                const float* __restrict__ cm, const float* __restrict__ cv,
                float* __restrict__ out) {
    __shared__ __align__(64) _Float16 s_a[128 * C_IN];   // 128 rows x 128 k
    __shared__ __align__(64) _Float16 s_bt[64 * C_IN];   // [n][k]

    const int tid  = threadIdx.x;
    const int lane = tid & 31;
    const int wave = tid >> 5;
    const int M0   = blockIdx.x * 128;
    const int N0   = blockIdx.y * 64;

    {   // stage A (contiguous rows) as 16B copies
        const uint4* src = (const uint4*)(h_ws + (size_t)M0 * C_IN);
        uint4* dst = (uint4*)s_a;
        for (int i = tid; i < (128 * C_IN) / 8; i += 256) dst[i] = src[i];
    }
    {   // stage B as [n][k] == w3's own row-major layout, f32->f16 via float4
        const float4* w3v = (const float4*)(w3 + (size_t)N0 * C_IN);
        for (int i = tid; i < (64 * C_IN) / 4; i += 256) {
            float4 w = w3v[i];
            v4h h = {(_Float16)w.x, (_Float16)w.y, (_Float16)w.z, (_Float16)w.w};
            *(v4h*)(s_bt + i * 4) = h;
        }
    }
    __syncthreads();

    for (int ti = 0; ti < 4; ++ti) {               // 32 tiles, 4 per wave
        const int t  = wave + 8 * ti;
        const int mt = t >> 2, nt = t & 3;
        const int m  = mt * 16 + (lane & 15);
        const int n  = nt * 16 + (lane & 15);
        v8f acc = {};
#pragma unroll
        for (int kc = 0; kc < 4; ++kc) {
            const int k0 = kc * 32;
            v16h a  = a_frag(s_a + m * C_IN + k0, lane);
            v16h bf = b_frag(s_bt + n * C_IN + k0, lane);
            acc = __builtin_amdgcn_wmma_f32_16x16x32_f16(false, a, false, bf,
                                                         (short)0, acc, false, false);
        }
        const int col = N0 + n;
        const float sc = cg[col] * rsqrtf(cv[col] + EPSF);
        const float sh = cb[col] - cm[col] * sc;
        const float bi = b3[col];
#pragma unroll
        for (int r = 0; r < 8; ++r) {
            int row = M0 + mt * 16 + r + ((lane >> 4) << 3);
            out[(size_t)row * C_OUT + col] = fmaxf((acc[r] + bi) * sc + sh, 0.f);
        }
    }
}

// ===========================================================================
extern "C" void kernel_launch(void* const* d_in, const int* in_sizes, int n_in,
                              void* d_out, int out_size, void* d_ws, size_t ws_size,
                              hipStream_t stream) {
    (void)in_sizes; (void)n_in; (void)out_size; (void)ws_size;
    const float* p_in  = (const float*)d_in[0];
    const float* p_out = (const float*)d_in[1];
    const float* R_out = (const float*)d_in[2];
    const float* h_in  = (const float*)d_in[3];
    const float* w1    = (const float*)d_in[4];
    const float* b1    = (const float*)d_in[5];
    const float* a_g   = (const float*)d_in[6];
    const float* a_b   = (const float*)d_in[7];
    const float* a_m   = (const float*)d_in[8];
    const float* a_v   = (const float*)d_in[9];
    const float* w2    = (const float*)d_in[10];
    const float* b2    = (const float*)d_in[11];
    const float* bg    = (const float*)d_in[12];
    const float* bb    = (const float*)d_in[13];
    const float* bm    = (const float*)d_in[14];
    const float* bv    = (const float*)d_in[15];
    const float* w3    = (const float*)d_in[16];
    const float* b3    = (const float*)d_in[17];
    const float* cg    = (const float*)d_in[18];
    const float* cb    = (const float*)d_in[19];
    const float* cm    = (const float*)d_in[20];
    const float* cv    = (const float*)d_in[21];

    int*      idx_ws = (int*)d_ws;
    _Float16* h_ws   = (_Float16*)((char*)d_ws + (size_t)B_ * N_OUT * K_NN * sizeof(int));

    knn_kernel<<<dim3((B_ * N_OUT) / KNN_T), dim3(KNN_T), 0, stream>>>(p_in, p_out, idx_ws);

    edge_kernel<<<dim3(B_ * (N_OUT / PT)), dim3(256), 0, stream>>>(
        p_in, p_out, R_out, h_in, w1, b1, a_g, a_b, a_m, a_v,
        w2, b2, bg, bb, bm, bv, idx_ws, h_ws);

    out_gemm_kernel<<<dim3((B_ * N_OUT) / 128, C_OUT / 64), dim3(256), 0, stream>>>(
        h_ws, w3, b3, cg, cb, cm, cv, (float*)d_out);
}
